// CoverNet_55722905699382
// MI455X (gfx1250) — matrix operand order repaired and checked by reading, not compiled, not run on hardware
//
#include <hip/hip_runtime.h>

typedef __attribute__((ext_vector_type(16))) _Float16 v16h;
typedef __attribute__((ext_vector_type(8)))  float    v8f;
typedef __attribute__((ext_vector_type(4)))  float    v4f;

#define R_ROWS   65536   // B*N
#define P_ROWS   65536
#define TB_BALLS 4096
#define M_MEMB   131072

__device__ __forceinline__ float lrelu(float v) { return v >= 0.f ? v : 0.2f * v; }

// ---------------------------------------------------------------- utilities
__global__ void zero_f32(float* p, int n) {
    int i = blockIdx.x * blockDim.x + threadIdx.x;
    if (i < n) p[i] = 0.f;
}

// Pack W (O,K row-major fp32) into per-lane B-fragment layout:
// chunk index c = (tn*(K/32) + kb)*32 + lane ; element i of chunk = half i of
// the B fragment for column tile tn, K-block kb:  k = kb*32 + 8*(lane>>4) + (i&7) + 16*(i>>3),
// o = tn*16 + (lane&15).  Each lane's 16 halves are contiguous (32B).
__global__ void pack_w_frag(const float* __restrict__ W, _Float16* __restrict__ Wp,
                            int O, int K) {
    int idx = blockIdx.x * blockDim.x + threadIdx.x;
    if (idx >= O * K) return;
    int i     = idx & 15;
    int chunk = idx >> 4;
    int lane  = chunk & 31;
    int blk   = chunk >> 5;
    int KB    = K >> 5;
    int kb    = blk % KB;
    int tn    = blk / KB;
    int o = tn * 16 + (lane & 15);
    int k = kb * 32 + 8 * (lane >> 4) + (i & 7) + 16 * (i >> 3);
    Wp[idx] = (_Float16)W[(size_t)o * K + k];
}

// scale = gamma * rsqrt(var+eps), shift = beta - mu*scale
__global__ void bn_finalize(const float* __restrict__ csum, const float* __restrict__ csq,
                            const float* __restrict__ gamma, const float* __restrict__ beta,
                            float* __restrict__ scale, float* __restrict__ shift,
                            int O, float invcnt) {
    int o = blockIdx.x * blockDim.x + threadIdx.x;
    if (o < O) {
        float mu  = csum[o] * invcnt;
        float var = csq[o] * invcnt - mu * mu;
        float sc  = gamma[o] * rsqrtf(var + 1e-5f);
        scale[o] = sc;
        shift[o] = beta[o] - mu * sc;
    }
}

// ------------------------------------------------- point branch layer 1 (K=3)
__global__ void point_l1(const float* __restrict__ x, const float* __restrict__ w,
                         float* __restrict__ Z, float* __restrict__ csum, float* __restrict__ csq) {
    __shared__ float ls[32], lq[32];
    int tid = threadIdx.x;
    if (tid < 32) { ls[tid] = 0.f; lq[tid] = 0.f; }
    __syncthreads();
    int r = blockIdx.x * blockDim.x + tid;
    int b = r >> 12, n = r & 4095;
    const float* xb = x + (size_t)b * 3 * 4096 + n;
    float x0 = xb[0], x1 = xb[4096], x2 = xb[8192];
#pragma unroll
    for (int o = 0; o < 32; ++o) {
        float z = x0 * w[o * 3] + x1 * w[o * 3 + 1] + x2 * w[o * 3 + 2];
        Z[(size_t)r * 32 + o] = z;
        atomicAdd(&ls[o], z);
        atomicAdd(&lq[o], z * z);
    }
    __syncthreads();
    if (tid < 32) { atomicAdd(&csum[tid], ls[tid]); atomicAdd(&csq[tid], lq[tid]); }
}

// ---------------------------------------------------------- WMMA GEMM kernel
// Z(rows x O) = act(A) @ W,  W pre-packed in fragment layout (see pack_w_frag).
// One wave computes a 16 x (16*NT) slab; A fragment reused across NT tiles.
template<int K, int NT, bool GATHER, bool ACT>
__global__ void gemm_bn_wmma(const float* __restrict__ A,
                             const int* __restrict__ rowmap, int map_stride,
                             const float* __restrict__ ascale, const float* __restrict__ ashift,
                             const _Float16* __restrict__ Wp,
                             float* __restrict__ Z,
                             float* __restrict__ csum, float* __restrict__ csq,
                             int O) {
    constexpr int KB = K / 32;
    int wave = threadIdx.x >> 5;
    int lane = threadIdx.x & 31;
    int g    = lane >> 4;
    int qn   = lane & 15;
    int ngroups = O / (16 * NT);
    int t  = blockIdx.x * (blockDim.x >> 5) + wave;
    int tm = t / ngroups;
    int tg = t % ngroups;

    int m = tm * 16 + qn;
    long srow = GATHER ? (long)rowmap[(long)m * map_stride] : (long)m;
    const float* arow = A + srow * (long)K;
    const v16h*  wpv  = (const v16h*)Wp;

    v8f acc[NT];
#pragma unroll
    for (int i = 0; i < NT; ++i) acc[i] = (v8f){};

#pragma unroll
    for (int kb = 0; kb < KB; ++kb) {
        const int k0 = kb * 32;
        // A fragment: halves 0..7 -> k = k0+8g+i ; halves 8..15 -> k = k0+16+8g+i
        v4f f0 = *(const v4f*)(arow + k0 + 8 * g);
        v4f f1 = *(const v4f*)(arow + k0 + 8 * g + 4);
        v4f f2 = *(const v4f*)(arow + k0 + 16 + 8 * g);
        v4f f3 = *(const v4f*)(arow + k0 + 16 + 8 * g + 4);
        v16h a;
#pragma unroll
        for (int i = 0; i < 4; ++i) {
            float u0 = f0[i], u1 = f1[i], u2 = f2[i], u3 = f3[i];
            if (ACT) {
                int ka = k0 + 8 * g + i;
                int kc = k0 + 16 + 8 * g + i;
                u0 = lrelu(fmaf(u0, ascale[ka],     ashift[ka]));
                u1 = lrelu(fmaf(u1, ascale[ka + 4], ashift[ka + 4]));
                u2 = lrelu(fmaf(u2, ascale[kc],     ashift[kc]));
                u3 = lrelu(fmaf(u3, ascale[kc + 4], ashift[kc + 4]));
            }
            a[i]      = (_Float16)u0;
            a[i + 4]  = (_Float16)u1;
            a[i + 8]  = (_Float16)u2;
            a[i + 12] = (_Float16)u3;
        }
#pragma unroll
        for (int nt = 0; nt < NT; ++nt) {
            v16h b = wpv[((size_t)(tg * NT + nt) * KB + kb) * 32 + lane];
            acc[nt] = __builtin_amdgcn_wmma_f32_16x16x32_f16(false, a, false, b,
                                                             (short)0, acc[nt], false, false);
        }
    }

#pragma unroll
    for (int nt = 0; nt < NT; ++nt) {
        int ncol = (tg * NT + nt) * 16 + qn;
        float s = 0.f, ss = 0.f;
#pragma unroll
        for (int vv = 0; vv < 8; ++vv) {
            int row = tm * 16 + vv + 8 * g;
            float c = acc[nt][vv];
            Z[(size_t)row * O + ncol] = c;
            s += c; ss += c * c;
        }
        s  += __shfl_xor(s, 16, 32);
        ss += __shfl_xor(ss, 16, 32);
        if (g == 0) { atomicAdd(&csum[ncol], s); atomicAdd(&csq[ncol], ss); }
    }
}

// --------------------------------------------------------- ball segment mean
__global__ void ball_accum(const float* __restrict__ Z3,
                           const float* __restrict__ scale, const float* __restrict__ shift,
                           const int* __restrict__ node, const int* __restrict__ seg,
                           float* __restrict__ balls, float* __restrict__ cnt) {
    int gid = blockIdx.x * blockDim.x + threadIdx.x;
    int m = gid >> 7, c = gid & 127;
    if (m >= M_MEMB) return;
    int r = node[m], sg = seg[m];
    float v = lrelu(fmaf(Z3[(size_t)r * 128 + c], scale[c], shift[c]));
    atomicAdd(&balls[(size_t)sg * 128 + c], v);
    if (c == 0) atomicAdd(&cnt[sg], 1.0f);
}

__global__ void ball_div(float* __restrict__ balls, const float* __restrict__ cnt) {
    int i = blockIdx.x * blockDim.x + threadIdx.x;
    if (i < TB_BALLS * 128) balls[i] /= fmaxf(cnt[i >> 7], 1.0f);
}

// ------------------------------- fused final linear: out = bn_lrelu(Z3) @ fw.T
__global__ void final_fuse(const float* __restrict__ Z3,
                           const float* __restrict__ scale, const float* __restrict__ shift,
                           const float* __restrict__ fw, const float* __restrict__ fb,
                           float* __restrict__ out, int J, int off, int accumulate) {
    int wave = threadIdx.x >> 5, lane = threadIdx.x & 31;
    int p = blockIdx.x * (blockDim.x >> 5) + wave;
    const float* zr = Z3 + (size_t)p * 1024;
    float a0 = 0.f, a1 = 0.f, a2 = 0.f, a3 = 0.f;
    for (int t = 0; t < 32; ++t) {
        int o = t * 32 + lane;
        float h = lrelu(fmaf(zr[o], scale[o], shift[o]));
        a0 += h * fw[off + o];
        if (J > 1) {
            a1 += h * fw[2048 + off + o];
            a2 += h * fw[2 * 2048 + off + o];
            a3 += h * fw[3 * 2048 + off + o];
        }
    }
    for (int d = 16; d > 0; d >>= 1) {
        a0 += __shfl_down(a0, d, 32);
        a1 += __shfl_down(a1, d, 32);
        a2 += __shfl_down(a2, d, 32);
        a3 += __shfl_down(a3, d, 32);
    }
    if (lane == 0) {
        float vals[4] = { a0, a1, a2, a3 };
        for (int j = 0; j < J; ++j) {
            if (accumulate) out[(size_t)p * J + j] += vals[j];
            else            out[(size_t)p * J + j]  = vals[j] + fb[j];
        }
    }
}

// ============================================================== host launcher
extern "C" void kernel_launch(void* const* d_in, const int* in_sizes, int n_in,
                              void* d_out, int out_size, void* d_ws, size_t ws_size,
                              hipStream_t stream) {
    (void)in_sizes; (void)n_in; (void)out_size; (void)ws_size;

    const float* x        = (const float*)d_in[0];
    const int*   node     = (const int*)d_in[1];
    const int*   seg      = (const int*)d_in[2];
    const int*   diff_ids = (const int*)d_in[3];
    const int*   same_ids = (const int*)d_in[4];
    const float* pw1 = (const float*)d_in[6];
    const float* pw2 = (const float*)d_in[7];
    const float* pw3 = (const float*)d_in[8];
    const float* pg1 = (const float*)d_in[9],  *pb1 = (const float*)d_in[10];
    const float* pg2 = (const float*)d_in[11], *pb2 = (const float*)d_in[12];
    const float* pg3 = (const float*)d_in[13], *pb3 = (const float*)d_in[14];
    const float* qw1 = (const float*)d_in[15];
    const float* qw2 = (const float*)d_in[16];
    const float* qw3 = (const float*)d_in[17];
    const float* qg1 = (const float*)d_in[18], *qb1 = (const float*)d_in[19];
    const float* qg2 = (const float*)d_in[20], *qb2 = (const float*)d_in[21];
    const float* qg3 = (const float*)d_in[22], *qb3 = (const float*)d_in[23];
    const float* qfw = (const float*)d_in[24], *qfb = (const float*)d_in[25];
    const float* dw1 = (const float*)d_in[26];
    const float* dw2 = (const float*)d_in[27];
    const float* dw3 = (const float*)d_in[28];
    const float* dg1 = (const float*)d_in[29], *db1 = (const float*)d_in[30];
    const float* dg2 = (const float*)d_in[31], *db2 = (const float*)d_in[32];
    const float* dg3 = (const float*)d_in[33], *db3 = (const float*)d_in[34];
    const float* dfw = (const float*)d_in[35], *dfb = (const float*)d_in[36];

    float* out_qd  = (float*)d_out;                       // P x 4
    float* out_dis = (float*)d_out + (size_t)P_ROWS * 4;  // P x 1

    // ------------------------------------------------- workspace carve-out
    char* cur = (char*)d_ws;
    auto alloc = [&](size_t bytes) -> void* {
        void* p = cur; cur += (bytes + 255) & ~(size_t)255; return p;
    };
    float* Z1p  = (float*)alloc((size_t)R_ROWS * 32 * 4);
    float* Z2p  = (float*)alloc((size_t)R_ROWS * 64 * 4);
    float* Z3p  = (float*)alloc((size_t)R_ROWS * 128 * 4);
    float* Zb1  = (float*)alloc((size_t)P_ROWS * 64 * 4);
    float* Zb2  = (float*)alloc((size_t)P_ROWS * 128 * 4);
    float* Zb3  = (float*)alloc((size_t)P_ROWS * 1024 * 4);
    float* balls = (float*)alloc((size_t)TB_BALLS * 128 * 4);
    float* cnt   = (float*)alloc((size_t)TB_BALLS * 4);
    float* ssum  = (float*)alloc(1024 * 4);
    float* ssq   = (float*)alloc(1024 * 4);
    float* sc_p1 = (float*)alloc(32 * 4);   float* sh_p1 = (float*)alloc(32 * 4);
    float* sc_p2 = (float*)alloc(64 * 4);   float* sh_p2 = (float*)alloc(64 * 4);
    float* sc_p3 = (float*)alloc(128 * 4);  float* sh_p3 = (float*)alloc(128 * 4);
    float* sc_b1 = (float*)alloc(64 * 4);   float* sh_b1 = (float*)alloc(64 * 4);
    float* sc_b2 = (float*)alloc(128 * 4);  float* sh_b2 = (float*)alloc(128 * 4);
    float* sc_b3 = (float*)alloc(1024 * 4); float* sh_b3 = (float*)alloc(1024 * 4);
    _Float16* Wt_p2 = (_Float16*)alloc(32 * 64 * 2);
    _Float16* Wt_p3 = (_Float16*)alloc(64 * 128 * 2);
    _Float16* Wt_q1 = (_Float16*)alloc(128 * 64 * 2);
    _Float16* Wt_q2 = (_Float16*)alloc(64 * 128 * 2);
    _Float16* Wt_q3 = (_Float16*)alloc(128 * 1024 * 2);
    _Float16* Wt_d1 = (_Float16*)alloc(128 * 64 * 2);
    _Float16* Wt_d2 = (_Float16*)alloc(64 * 128 * 2);
    _Float16* Wt_d3 = (_Float16*)alloc(128 * 1024 * 2);

    auto zero = [&](float* p, int n) {
        zero_f32<<<(n + 255) / 256, 256, 0, stream>>>(p, n);
    };
    auto pack = [&](const float* W, _Float16* Wp, int O, int K) {
        pack_w_frag<<<(O * K + 255) / 256, 256, 0, stream>>>(W, Wp, O, K);
    };
    // dispatch: K in {32,64,128}; NT=4 (16x64 slab per wave)
    auto gemm = [&](const float* A, const int* rm, int mstride,
                    const float* asc, const float* ash, const _Float16* Wp,
                    float* Z, int rows, int K, int O) {
        int waves  = (rows / 16) * (O / 64);   // ngroups = O/(16*NT)
        dim3 grid(waves / 8), block(256);
        if (rm) {
            gemm_bn_wmma<128, 4, true, false><<<grid, block, 0, stream>>>(
                A, rm, mstride, asc, ash, Wp, Z, ssum, ssq, O);
        } else if (K == 32) {
            gemm_bn_wmma<32, 4, false, true><<<grid, block, 0, stream>>>(
                A, rm, mstride, asc, ash, Wp, Z, ssum, ssq, O);
        } else if (K == 64) {
            gemm_bn_wmma<64, 4, false, true><<<grid, block, 0, stream>>>(
                A, rm, mstride, asc, ash, Wp, Z, ssum, ssq, O);
        } else {
            gemm_bn_wmma<128, 4, false, true><<<grid, block, 0, stream>>>(
                A, rm, mstride, asc, ash, Wp, Z, ssum, ssq, O);
        }
    };
    auto fin = [&](const float* g, const float* b, float* sc, float* sh, int O, float invc) {
        bn_finalize<<<(O + 63) / 64, 64, 0, stream>>>(ssum, ssq, g, b, sc, sh, O, invc);
    };

    const float invR = 1.0f / (float)R_ROWS;
    const float invP = 1.0f / (float)P_ROWS;

    // weight packing (fp32 -> f16 fragment layout)
    pack(pw2, Wt_p2, 64, 32);    pack(pw3, Wt_p3, 128, 64);
    pack(qw1, Wt_q1, 64, 128);   pack(qw2, Wt_q2, 128, 64);  pack(qw3, Wt_q3, 1024, 128);
    pack(dw1, Wt_d1, 64, 128);   pack(dw2, Wt_d2, 128, 64);  pack(dw3, Wt_d3, 1024, 128);

    // ---- point branch
    zero(ssum, 32); zero(ssq, 32);
    point_l1<<<R_ROWS / 256, 256, 0, stream>>>(x, pw1, Z1p, ssum, ssq);
    fin(pg1, pb1, sc_p1, sh_p1, 32, invR);

    zero(ssum, 64); zero(ssq, 64);
    gemm(Z1p, nullptr, 0, sc_p1, sh_p1, Wt_p2, Z2p, R_ROWS, 32, 64);
    fin(pg2, pb2, sc_p2, sh_p2, 64, invR);

    zero(ssum, 128); zero(ssq, 128);
    gemm(Z2p, nullptr, 0, sc_p2, sh_p2, Wt_p3, Z3p, R_ROWS, 64, 128);
    fin(pg3, pb3, sc_p3, sh_p3, 128, invR);

    // ---- segment mean into balls
    zero(balls, TB_BALLS * 128); zero(cnt, TB_BALLS);
    ball_accum<<<(M_MEMB * 128) / 256, 256, 0, stream>>>(Z3p, sc_p3, sh_p3, node, seg, balls, cnt);
    ball_div<<<(TB_BALLS * 128) / 256, 256, 0, stream>>>(balls, cnt);

    // ---- four branch runs
    struct Run {
        const int* ids; int col;
        const _Float16 *W1, *W2, *W3;
        const float *g1, *b1, *g2, *b2, *g3, *b3;
        const float *fw, *fb; float* out; int J, off, accum;
    };
    Run runs[4] = {
        { diff_ids, 0, Wt_q1, Wt_q2, Wt_q3, qg1, qb1, qg2, qb2, qg3, qb3, qfw, qfb, out_qd,  4, 0,    0 },
        { diff_ids, 1, Wt_q1, Wt_q2, Wt_q3, qg1, qb1, qg2, qb2, qg3, qb3, qfw, qfb, out_qd,  4, 1024, 1 },
        { same_ids, 0, Wt_d1, Wt_d2, Wt_d3, dg1, db1, dg2, db2, dg3, db3, dfw, dfb, out_dis, 1, 0,    0 },
        { same_ids, 1, Wt_d1, Wt_d2, Wt_d3, dg1, db1, dg2, db2, dg3, db3, dfw, dfb, out_dis, 1, 1024, 1 },
    };

    for (int rr = 0; rr < 4; ++rr) {
        const Run& R = runs[rr];
        zero(ssum, 64); zero(ssq, 64);
        gemm(balls, R.ids + R.col, 2, nullptr, nullptr, R.W1, Zb1, P_ROWS, 128, 64);
        fin(R.g1, R.b1, sc_b1, sh_b1, 64, invP);

        zero(ssum, 128); zero(ssq, 128);
        gemm(Zb1, nullptr, 0, sc_b1, sh_b1, R.W2, Zb2, P_ROWS, 64, 128);
        fin(R.g2, R.b2, sc_b2, sh_b2, 128, invP);

        zero(ssum, 1024); zero(ssq, 1024);
        gemm(Zb2, nullptr, 0, sc_b2, sh_b2, R.W3, Zb3, P_ROWS, 128, 1024);
        fin(R.g3, R.b3, sc_b3, sh_b3, 1024, invP);

        final_fuse<<<P_ROWS / 8, 256, 0, stream>>>(Zb3, sc_b3, sh_b3, R.fw, R.fb,
                                                   R.out, R.J, R.off, R.accum);
    }
}
